// SingleLevelDiVeQ_69647189672429
// MI455X (gfx1250) — matrix operand order repaired and checked by reading, not compiled, not run on hardware
//
#include <hip/hip_runtime.h>
#include <hip/hip_bf16.h>

// ---------------------------------------------------------------------------
// Problem constants (from reference): B=32768, D_IN=1024, D_Q=512, K=8192
// ---------------------------------------------------------------------------
#define NB      32768
#define D_IN    1024
#define D_Q     512
#define KCB     8192

typedef __attribute__((ext_vector_type(16))) __bf16 v16bf;
typedef __attribute__((ext_vector_type(8)))  __bf16 v8bf;
typedef __attribute__((ext_vector_type(8)))  float  v8f;

// ---------------------------------------------------------------------------
// CDNA5 async global->LDS copy (CDNA5 ISA).
// GVS mode: mem_addr = SADDR(64b SGPR) + VADDR(32b VGPR) ; VDST = LDS byte
// offset (hardware adds the wave's LDS_BASE). Tracked by ASYNCcnt.
// ---------------------------------------------------------------------------
__device__ __forceinline__ void async_ld_b128(unsigned lds_byte_off,
                                              unsigned g_byte_off,
                                              unsigned long long g_base) {
  asm volatile("global_load_async_to_lds_b128 %0, %1, %2"
               :: "v"(lds_byte_off), "v"(g_byte_off), "s"(g_base)
               : "memory");
}
__device__ __forceinline__ void wait_async0() {
  asm volatile("s_wait_asynccnt 0" ::: "memory");
}
__device__ __forceinline__ unsigned lds_off(const void* p) {
  return (unsigned)(unsigned long long)p;  // flat addr[31:0] == LDS offset
}

// ---------------------------------------------------------------------------
// CDNA5 WMMA fragment helpers (this code's contiguous k order k = 16*hi + s, same for A and B)
// A 16x32 bf16: lane l (r=l&15, hi=l>>4) holds row r,
//   halves [hi*16 .. hi*16+7] and [hi*16+8 .. hi*16+15] -> two 16B LDS reads.
// B 32x16 bf16 (stored N-major, K-contiguous "Bt"): lane l holds col l&15,
//   halves [hi*16 .. hi*16+15] -> one 32B LDS read.
// ---------------------------------------------------------------------------
__device__ __forceinline__ v16bf load_a_frag(const __bf16* rowptr, int hi) {
  v8bf lo = *(const v8bf*)(rowptr + hi * 16);
  v8bf hh = *(const v8bf*)(rowptr + hi * 16 + 8);
  return __builtin_shufflevector(lo, hh, 0,1,2,3,4,5,6,7,8,9,10,11,12,13,14,15);
}

__device__ __forceinline__ v8f wmma_bf16(v16bf a, v16bf b, v8f c) {
  return __builtin_amdgcn_wmma_f32_16x16x32_bf16(false, a, false, b,
                                                 (short)0, c, false, false);
}

// ---------------------------------------------------------------------------
// Prep kernels
// ---------------------------------------------------------------------------
__global__ __launch_bounds__(256)
void cvt_bf16_kernel(const float* __restrict__ in, __bf16* __restrict__ out,
                     long long n) {
  long long i = ((long long)blockIdx.x * 256 + threadIdx.x) * 4;
  if (i + 3 < n) {
    float4 v = *(const float4*)(in + i);
    out[i + 0] = (__bf16)v.x;
    out[i + 1] = (__bf16)v.y;
    out[i + 2] = (__bf16)v.z;
    out[i + 3] = (__bf16)v.w;
  }
}

// out[c][r] = (bf16) in[r][c]   (make weights K-contiguous for WMMA B operand)
__global__ void transpose_cvt_kernel(const float* __restrict__ in,
                                     __bf16* __restrict__ out, int R, int C) {
  int c = blockIdx.x * 16 + threadIdx.x;
  int r = blockIdx.y * 16 + threadIdx.y;
  if (r < R && c < C)
    out[(long long)c * R + r] = (__bf16)in[(long long)r * C + c];
}

// codebook row -> bf16 copy + sum of squares
__global__ __launch_bounds__(256)
void cb_prep_kernel(const float* __restrict__ cb, __bf16* __restrict__ cbb,
                    float* __restrict__ cbsq) {
  const int k = blockIdx.x;
  const int t = threadIdx.x;
  float s = 0.f;
  #pragma unroll
  for (int i = t; i < D_Q; i += 256) {
    float v = cb[(long long)k * D_Q + i];
    cbb[(long long)k * D_Q + i] = (__bf16)v;
    s += v * v;
  }
  #pragma unroll
  for (int m = 16; m >= 1; m >>= 1) s += __shfl_xor(s, m, 32);
  __shared__ float red[8];
  if ((t & 31) == 0) red[t >> 5] = s;
  __syncthreads();
  if (t < 8) {
    float v = red[t];
    #pragma unroll
    for (int m = 4; m >= 1; m >>= 1) v += __shfl_xor(v, m, 32);
    if (t == 0) cbsq[k] = v;
  }
}

// ---------------------------------------------------------------------------
// Generic tiled WMMA GEMM:  C[M,N] = A[M,K] * Bt[N,K]^T + bias[N]
// Block = 256 threads = 8 waves; block tile 128(M) x 64(N); K-tile 64.
// Async double-buffered staging: fill tile kt+1 while computing tile kt.
// ---------------------------------------------------------------------------
#define GA_STRIDE 80   // 64 halves + pad, rows 16B-aligned (160 B)
#define GB_STRIDE 80

template <bool OUT_BF16>
__global__ __launch_bounds__(256)
void gemm_wmma_bf16(const __bf16* __restrict__ A, const __bf16* __restrict__ Bt,
                    const float* __restrict__ bias, float* __restrict__ Cf,
                    __bf16* __restrict__ Cb, int Kdim, int N) {
  __shared__ __bf16 sA[2 * 128 * GA_STRIDE];   // 40 KB
  __shared__ __bf16 sB[2 * 64 * GB_STRIDE];    // 20 KB

  const int tid  = threadIdx.x;
  const int wave = tid >> 5;
  const int lane = tid & 31;
  const int ln   = lane & 15;
  const int hi   = lane >> 4;
  const long long mbase = (long long)blockIdx.x * 128;
  const int nbase = blockIdx.y * 64;

  const unsigned long long baseA = (unsigned long long)A;
  const unsigned long long baseB = (unsigned long long)Bt;

  // async-stage one K-tile into buffer `buf`
  auto stage = [&](int k0, int buf) {
    __bf16* dA = sA + buf * 128 * GA_STRIDE;
    __bf16* dB = sB + buf * 64 * GB_STRIDE;
    #pragma unroll
    for (int i = 0; i < 4; ++i) {           // A: 1024 x 16B chunks
      int id = tid + i * 256;
      int r = id >> 3, c = id & 7;
      async_ld_b128(lds_off(&dA[r * GA_STRIDE + c * 8]),
                    (unsigned)(((mbase + r) * Kdim + k0 + c * 8) * 2), baseA);
    }
    #pragma unroll
    for (int i = 0; i < 2; ++i) {           // Bt: 512 x 16B chunks
      int id = tid + i * 256;
      int r = id >> 3, c = id & 7;
      async_ld_b128(lds_off(&dB[r * GB_STRIDE + c * 8]),
                    (unsigned)(((long long)(nbase + r) * Kdim + k0 + c * 8) * 2),
                    baseB);
    }
  };

  v8f acc[4] = {};
  const int ktiles = Kdim >> 6;

  stage(0, 0);
  wait_async0();
  __syncthreads();

  for (int kt = 0; kt < ktiles; ++kt) {
    const int cur = kt & 1;
    if (kt + 1 < ktiles) stage((kt + 1) << 6, cur ^ 1);

    const __bf16* cA = sA + cur * 128 * GA_STRIDE;
    const __bf16* cB = sB + cur * 64 * GB_STRIDE;
    #pragma unroll
    for (int ks = 0; ks < 2; ++ks) {
      v16bf afrag = load_a_frag(&cA[(wave * 16 + ln) * GA_STRIDE + ks * 32], hi);
      #pragma unroll
      for (int ns = 0; ns < 4; ++ns) {
        v16bf bfrag =
            *(const v16bf*)(&cB[(ns * 16 + ln) * GB_STRIDE + ks * 32 + hi * 16]);
        acc[ns] = wmma_bf16(afrag, bfrag, acc[ns]);
      }
    }
    wait_async0();     // next tile resident (this wave's copies)
    __syncthreads();   // all waves done computing cur + all fills visible
  }

  // Epilogue: acc[ns][g] holds C(m = g + 8*hi, n = ln) of subtile ns
  #pragma unroll
  for (int ns = 0; ns < 4; ++ns) {
    #pragma unroll
    for (int g = 0; g < 8; ++g) {
      int n = nbase + ns * 16 + ln;
      long long m = mbase + wave * 16 + g + hi * 8;
      float v = acc[ns][g] + bias[n];
      if (OUT_BF16) Cb[m * N + n] = (__bf16)v;
      else          Cf[m * N + n] = v;
    }
  }
}

// ---------------------------------------------------------------------------
// Fused distance GEMM + argmin.
// dist(b,k) = ||c_k||^2 - 2 * <ze_q[b], c_k>  (||ze_q||^2 is row-constant).
// Block owns 128 rows; full K=512 of A staged once (async); codebook streamed
// 32 rows/iter through a double buffer filled with async-to-LDS copies.
// ---------------------------------------------------------------------------
#define SA_STRIDE 520   // mult of 8  -> 16B-aligned rows (1040 B)
#define SB_STRIDE 528   // mult of 16 -> 32B-aligned rows (1056 B)
#define DIST_LDS_BYTES (128 * SA_STRIDE * 2 + 2 * 32 * SB_STRIDE * 2)  // 200704

__global__ __launch_bounds__(256)
void dist_argmin_wmma(const __bf16* __restrict__ Zq,
                      const __bf16* __restrict__ Cbk,
                      const float* __restrict__ cbsq,
                      int* __restrict__ idx_out, float* __restrict__ idxf_out) {
  extern __shared__ char smem_raw[];
  __bf16* sA = (__bf16*)smem_raw;            // 128 x 512 (stride 520)
  __bf16* sB = sA + 128 * SA_STRIDE;         // 2 x [32 x 512] (stride 528)

  const int tid  = threadIdx.x;
  const int wave = tid >> 5;
  const int lane = tid & 31;
  const int ln   = lane & 15;
  const int hi   = lane >> 4;
  const long long mbase = (long long)blockIdx.x * 128;
  const unsigned long long baseZ = (unsigned long long)Zq;
  const unsigned long long baseC = (unsigned long long)Cbk;

  // async-stage 32 codebook rows into buffer `buf`
  auto stage_cb = [&](int nb, int buf) {
    __bf16* d = sB + buf * 32 * SB_STRIDE;
    #pragma unroll
    for (int i = 0; i < 8; ++i) {            // 2048 x 16B chunks
      int id = tid + i * 256;
      int r = id >> 6, c = id & 63;
      async_ld_b128(lds_off(&d[r * SB_STRIDE + c * 8]),
                    (unsigned)(((long long)(nb + r) * D_Q + c * 8) * 2), baseC);
    }
  };

  // Stage full A block (128 x 512 halves) asynchronously: 32 chunks/thread
  #pragma unroll
  for (int i = 0; i < 32; ++i) {
    int id = tid + i * 256;
    int r = id >> 6, c = id & 63;
    async_ld_b128(lds_off(&sA[r * SA_STRIDE + c * 8]),
                  (unsigned)(((mbase + r) * D_Q + c * 8) * 2), baseZ);
  }
  stage_cb(0, 0);
  wait_async0();
  __syncthreads();

  float best[8];
  int   bidx[8];
  #pragma unroll
  for (int g = 0; g < 8; ++g) { best[g] = 3.4e38f; bidx[g] = 0; }

  for (int nt = 0; nt < KCB / 32; ++nt) {
    const int nb = nt * 32;
    const int cur = nt & 1;
    if (nt + 1 < KCB / 32) stage_cb(nb + 32, cur ^ 1);

    const __bf16* cB = sB + cur * 32 * SB_STRIDE;
    v8f acc0 = {}, acc1 = {};
    #pragma unroll
    for (int kt = 0; kt < 16; ++kt) {
      const int k0 = kt * 32;
      v16bf afrag = load_a_frag(&sA[(wave * 16 + ln) * SA_STRIDE + k0], hi);
      v16bf b0 = *(const v16bf*)(&cB[ln * SB_STRIDE + k0 + hi * 16]);
      v16bf b1 = *(const v16bf*)(&cB[(16 + ln) * SB_STRIDE + k0 + hi * 16]);
      acc0 = wmma_bf16(afrag, b0, acc0);
      acc1 = wmma_bf16(afrag, b1, acc1);
    }

    const float q0 = cbsq[nb + ln];
    const float q1 = cbsq[nb + 16 + ln];
    #pragma unroll
    for (int g = 0; g < 8; ++g) {
      float d0 = q0 - 2.0f * acc0[g];
      float d1 = q1 - 2.0f * acc1[g];
      if (d0 < best[g]) { best[g] = d0; bidx[g] = nb + ln; }
      if (d1 < best[g]) { best[g] = d1; bidx[g] = nb + 16 + ln; }
    }
    wait_async0();     // next codebook tile resident (this wave)
    __syncthreads();   // all waves done with cur; fills visible to all
  }

  // Cross-lane reduce among the 16 lanes sharing each row (xor masks 1..8
  // stay within each 16-lane half of the wave32); tie-break -> smaller index
  #pragma unroll
  for (int m = 1; m < 16; m <<= 1) {
    #pragma unroll
    for (int g = 0; g < 8; ++g) {
      float ov = __shfl_xor(best[g], m, 32);
      int   oi = __shfl_xor(bidx[g], m, 32);
      if (ov < best[g] || (ov == best[g] && oi < bidx[g])) {
        best[g] = ov; bidx[g] = oi;
      }
    }
  }
  if (ln == 0) {
    #pragma unroll
    for (int g = 0; g < 8; ++g) {
      long long m = mbase + wave * 16 + g + hi * 8;
      idx_out[m]  = bidx[g];
      idxf_out[m] = (float)bidx[g];
    }
  }
}

// ---------------------------------------------------------------------------
// Gather: quantized_q[b] = codebook[idx[b]]  (f32 out + bf16 staging)
// ---------------------------------------------------------------------------
__global__ __launch_bounds__(256)
void gather_kernel(const float* __restrict__ cb, const int* __restrict__ idx,
                   float* __restrict__ qq_f, __bf16* __restrict__ qq_b) {
  const long long b = blockIdx.x;
  const int t = threadIdx.x;
  const long long k = idx[b];
  #pragma unroll
  for (int i = 0; i < 2; ++i) {
    int c = t + i * 256;
    float v = cb[k * D_Q + c];
    qq_f[b * D_Q + c] = v;
    qq_b[b * D_Q + c] = (__bf16)v;
  }
}

// ---------------------------------------------------------------------------
// Host launcher
// Inputs (setup_inputs order): ze, W_in, b_in, codebook, W_out, b_out
// Output (flat concat): idx_f32[32768] | quantized[32768*1024] | qq[32768*512]
// ---------------------------------------------------------------------------
extern "C" void kernel_launch(void* const* d_in, const int* in_sizes, int n_in,
                              void* d_out, int out_size, void* d_ws,
                              size_t ws_size, hipStream_t stream) {
  const float* ze    = (const float*)d_in[0];
  const float* W_in  = (const float*)d_in[1];
  const float* b_in  = (const float*)d_in[2];
  const float* cbk   = (const float*)d_in[3];
  const float* W_out = (const float*)d_in[4];
  const float* b_out = (const float*)d_in[5];

  float* out_idx_f = (float*)d_out;
  float* out_quant = out_idx_f + NB;
  float* out_qq    = out_quant + (long long)NB * D_IN;

  // Workspace layout (bytes, 256-aligned)
  char* ws = (char*)d_ws;
  size_t off = 0;
  auto take = [&](size_t bytes) {
    char* p = ws + off;
    off += (bytes + 255) & ~(size_t)255;
    return p;
  };
  __bf16* ze_b   = (__bf16*)take((size_t)NB * D_IN * 2);   // 64 MB
  __bf16* winT   = (__bf16*)take((size_t)D_Q * D_IN * 2);  // 1 MB  [512][1024]
  __bf16* woutT  = (__bf16*)take((size_t)D_IN * D_Q * 2);  // 1 MB  [1024][512]
  __bf16* cbk_b  = (__bf16*)take((size_t)KCB * D_Q * 2);   // 8 MB
  float*  cbsq   = (float*) take((size_t)KCB * 4);
  __bf16* zeq_b  = (__bf16*)take((size_t)NB * D_Q * 2);    // 32 MB
  __bf16* qq_b   = (__bf16*)take((size_t)NB * D_Q * 2);    // 32 MB
  int*    idx_i  = (int*)   take((size_t)NB * 4);
  (void)ws_size; (void)in_sizes; (void)n_in; (void)out_size;

  // 1) ze -> bf16
  {
    long long n = (long long)NB * D_IN;
    cvt_bf16_kernel<<<(unsigned)(n / 4 / 256), 256, 0, stream>>>(ze, ze_b, n);
  }
  // 2) W_in [1024,512] -> winT [512,1024] ; W_out [512,1024] -> woutT [1024,512]
  transpose_cvt_kernel<<<dim3(D_Q / 16, D_IN / 16), dim3(16, 16), 0, stream>>>(
      W_in, winT, D_IN, D_Q);
  transpose_cvt_kernel<<<dim3(D_IN / 16, D_Q / 16), dim3(16, 16), 0, stream>>>(
      W_out, woutT, D_Q, D_IN);
  // 3) codebook -> bf16 + row norms
  cb_prep_kernel<<<KCB, 256, 0, stream>>>(cbk, cbk_b, cbsq);

  // 4) GEMM1: ze_q = bf16(ze @ W_in + b_in)   [32768,512]
  gemm_wmma_bf16<true><<<dim3(NB / 128, D_Q / 64), 256, 0, stream>>>(
      ze_b, winT, b_in, nullptr, zeq_b, D_IN, D_Q);

  // 5) Fused distance + argmin -> indices
  dist_argmin_wmma<<<NB / 128, 256, DIST_LDS_BYTES, stream>>>(
      zeq_b, cbk_b, cbsq, idx_i, out_idx_f);

  // 6) Gather quantized_q = codebook[idx]  (f32 output + bf16 staging)
  gather_kernel<<<NB, 256, 0, stream>>>(cbk, idx_i, out_qq, qq_b);

  // 7) GEMM3: quantized = qq @ W_out + b_out   [32768,1024] f32
  gemm_wmma_bf16<false><<<dim3(NB / 128, D_IN / 64), 256, 0, stream>>>(
      qq_b, woutT, b_out, out_quant, nullptr, D_Q, D_IN);
}
